// QLSTM_65481071397825
// MI455X (gfx1250) — matrix operand order repaired and checked by reading, not compiled, not run on hardware
//
#include <hip/hip_runtime.h>

// ---------------- problem constants ----------------
#define T_STEPS 256
#define BATCH   1024
#define IN_DIM  256
#define HID     1024
#define NQ      64
#define DEPTHL  4
#define DTOT    1280          // IN_DIM + HID
#define NCOLS   256           // 4 gates * NQ

// ---------------- tiling ----------------
#define ROWS  16              // batch rows per workgroup (one 16-row WMMA M-tile)
#define NWG   (BATCH / ROWS)  // 64 workgroups -> 64 WGPs busy per step
#define NT    2               // N-tiles (16 cols) per wave: 8 waves * 32 cols = 256
#define CSTR  1288            // comb LDS row stride in bf16 elems (1280 + 8 pad -> bank stagger)
#define ZSTR  264             // z bf16 row stride (256 + 8 pad)
#define ZFSTR 260             // z f32 row stride (256 + 4 pad)

typedef __attribute__((ext_vector_type(16))) __bf16        v16bf;
typedef __attribute__((ext_vector_type(8)))  float          v8f;
typedef __attribute__((ext_vector_type(8)))  unsigned short ush8;
typedef unsigned short ush;

// LDS layout (bytes)
#define COMB_BYTES (ROWS * CSTR * 2)                 // 41216
#define ZB_BYTES   (ROWS * ZSTR * 2)                 // 8448
#define OFF_ZB0    COMB_BYTES
#define OFF_ZB1    (OFF_ZB0 + ZB_BYTES)
#define OFF_ZF     (OFF_ZB1 + ZB_BYTES)
#define SMEM_BYTES (OFF_ZF + ROWS * ZFSTR * 4)       // 74752 bytes

__device__ __forceinline__ float sigm(float x) { return 1.0f / (1.0f + __expf(-x)); }
__device__ __forceinline__ float tanh_f(float x) {
  float a = fabsf(x);
  float e = __expf(-2.0f * a);           // v_exp_f32 (TRANS), co-executes with WMMA
  float t = (1.0f - e) / (1.0f + e);
  return x < 0.0f ? -t : t;
}
__device__ __forceinline__ ush f2bf(float f) {       // round-to-nearest-even fp32 -> bf16
  unsigned u = __builtin_bit_cast(unsigned, f);
  u += 0x7FFFu + ((u >> 16) & 1u);
  return (ush)(u >> 16);
}
__device__ __forceinline__ v8f wmma_bf16(v16bf a, v16bf b, v8f c) {
  return __builtin_amdgcn_wmma_f32_16x16x32_bf16(false, a, false, b, (short)0, c, false, false);
}

// ---------------- prep: fp32 weights -> bf16 (once) ----------------
__global__ void qlstm_prep(const float* __restrict__ lin_w, const float* __restrict__ layers_w,
                           ush* __restrict__ Wbf, ush* __restrict__ Lbf) {
  int idx = blockIdx.x * blockDim.x + threadIdx.x;
  int stride = gridDim.x * blockDim.x;
  const int NW = NCOLS * DTOT;          // 327680
  const int NL = 4 * DEPTHL * NQ * NQ;  // 65536
  for (int i = idx; i < NW; i += stride) Wbf[i] = f2bf(lin_w[i]);
  for (int i = idx; i < NL; i += stride) Lbf[i] = f2bf(layers_w[i]);
}

// ---------------- fused per-timestep kernel ----------------
// Phase 1: elementwise LSTM update for step t-1 (gates index h!), writes outs[t-1]/cx,
//          keeps hx as bf16 in LDS (the A-operand rows of this step's GEMM).
// Phase 2: z = comb @ W^T via v_wmma_f32_16x16x32_bf16, K=1280 in 40 chunks.
// Phase 3: 4 block-diagonal depth layers (per-gate 64x64) via WMMA + tanh, LDS ping-pong.
// Phase 4: readout -> gate logits for step t (consumed by kernel t+1).
__global__ __launch_bounds__(256) void qlstm_step(
    const float* __restrict__ inputs, const ush* __restrict__ Wbf, const ush* __restrict__ Lbf,
    const float* __restrict__ lin_b, const float* __restrict__ layers_b,
    const float* __restrict__ ro_w, const float* __restrict__ ro_b,
    const float* __restrict__ gprev, float* __restrict__ gnext,
    float* __restrict__ cx, float* __restrict__ outs, int t)
{
  extern __shared__ char smem[];
  ush*   comb = (ush*)smem;                    // [ROWS][CSTR] bf16: cols 0..255 = x, 256..1279 = hx
  ush*   zb0  = (ush*)(smem + OFF_ZB0);        // [ROWS][ZSTR] bf16
  ush*   zb1  = (ush*)(smem + OFF_ZB1);
  float* zf   = (float*)(smem + OFF_ZF);       // [ROWS][ZFSTR] f32 (final layer); aliases gate tables
  float* gF   = zf;                            // phase-1 only, dead before zf is written
  float* gIG  = zf + HID;
  float* gO   = zf + 2 * HID;

  const int tid  = threadIdx.x;
  const int wg   = blockIdx.x;
  const int row0 = wg * ROWS;

  // ---- phase 1: elementwise update (step t-1) + hx -> LDS ----
  if (t > 0) {
    for (int h = tid; h < HID; h += 256) {
      float f = sigm(gprev[h]);
      float i = sigm(gprev[HID + h]);
      float u = tanh_f(gprev[2 * HID + h]);
      float o = sigm(gprev[3 * HID + h]);
      gF[h] = f; gIG[h] = i * u; gO[h] = o;
    }
    __syncthreads();
    size_t ob = ((size_t)(t - 1) * BATCH + row0) * HID;
    for (int e = tid; e < ROWS * HID; e += 256) {
      int r = e >> 10, h = e & (HID - 1);
      size_t ci = (size_t)(row0 + r) * HID + h;
      float cold = (t == 1) ? 0.0f : cx[ci];
      float cn = gF[h] * cold + gIG[h];
      float hv = gO[h] * tanh_f(cn);
      cx[ci] = cn;
      outs[ob + (size_t)r * HID + h] = hv;
      comb[r * CSTR + IN_DIM + h] = f2bf(hv);
    }
  } else {
    for (int e = tid; e < ROWS * HID; e += 256) {
      int r = e >> 10, h = e & (HID - 1);
      comb[r * CSTR + IN_DIM + h] = 0;   // hx0 = 0
    }
  }
  // stage x[t] rows -> bf16 LDS; prefetch x[t+1] tile into cache for the next kernel
  {
    const float* xp = inputs + ((size_t)t * BATCH + row0) * IN_DIM;
    for (int e = tid; e < ROWS * IN_DIM; e += 256) {
      int r = e >> 8, d = e & (IN_DIM - 1);
      comb[r * CSTR + d] = f2bf(xp[r * IN_DIM + d]);
    }
    if (t + 1 < T_STEPS) {
      const char* nx = (const char*)(inputs + ((size_t)(t + 1) * BATCH + row0) * IN_DIM);
      __builtin_prefetch(nx + tid * 64, 0, 0);   // global_prefetch_b8; tile = 16KB = 256*64B
    }
  }
  __syncthreads();

  // ---- phase 2: main gate GEMM ----
  const int lane = tid & 31, wv = tid >> 5;
  const int lane15 = lane & 15, laneh = lane >> 4;
  const int aRow  = lane15;            // single 16-row M-tile
  const int mBase = laneh * 8;

  v8f acc[NT] = {};
  for (int kc = 0; kc < DTOT; kc += 32) {
    union { ush8 h[2]; v16bf v; } a;
    int kk = kc + 16 * laneh;                              // A: lanes 0-15 K[0..15], 16-31 K[16..31]
    a.h[0] = *(const ush8*)&comb[aRow * CSTR + kk];
    a.h[1] = *(const ush8*)&comb[aRow * CSTR + kk + 8];
    int krow = kc + 16 * laneh;                            // B: lanes 0-15 K[0..15], 16-31 K[16..31]
#pragma unroll
    for (int j = 0; j < NT; ++j) {
      int n = wv * 32 + j * 16 + lane15;
      v16bf b = *(const v16bf*)&Wbf[(size_t)n * DTOT + krow];
      acc[j] = wmma_bf16(a.v, b, acc[j]);
    }
  }
#pragma unroll
  for (int j = 0; j < NT; ++j) {
    int n = wv * 32 + j * 16 + lane15;
    float bias = lin_b[n];
#pragma unroll
    for (int r = 0; r < 8; ++r)
      zb0[(mBase + r) * ZSTR + n] = f2bf(acc[j][r] + bias);
  }

  // ---- phase 3: depth layers; this wave's 32-col strip lies inside gate g = wv>>1 ----
  const int g  = wv >> 1;
  const int p0 = (wv & 1) * 32;        // offset within the gate's 64-col block
  ush* zin = zb0; ush* zout = zb1;
  for (int l = 0; l < DEPTHL; ++l) {
    __syncthreads();
    v8f acc2[NT] = {};
#pragma unroll
    for (int kc = 0; kc < NQ; kc += 32) {
      union { ush8 h[2]; v16bf v; } a;
      int kk = kc + 16 * laneh;
      a.h[0] = *(const ush8*)&zin[aRow * ZSTR + g * NQ + kk];
      a.h[1] = *(const ush8*)&zin[aRow * ZSTR + g * NQ + kk + 8];
      int krow = kc + 16 * laneh;
#pragma unroll
      for (int j = 0; j < NT; ++j) {
        int p = p0 + j * 16 + lane15;
        v16bf b = *(const v16bf*)&Lbf[(((g * DEPTHL + l) * NQ + p) * NQ) + krow];
        acc2[j] = wmma_bf16(a.v, b, acc2[j]);
      }
    }
#pragma unroll
    for (int j = 0; j < NT; ++j) {
      int p = p0 + j * 16 + lane15;
      float bias = layers_b[(g * DEPTHL + l) * NQ + p];
#pragma unroll
      for (int r = 0; r < 8; ++r) {
        float v = tanh_f(acc2[j][r] + bias);
        if (l < DEPTHL - 1) zout[(mBase + r) * ZSTR + g * NQ + p] = f2bf(v);
        else                zf[(mBase + r) * ZFSTR + g * NQ + p] = v;
      }
    }
    ush* tmp = zin; zin = zout; zout = tmp;
  }
  __syncthreads();

  // ---- phase 4: readout -> gate logits for step t ----
  if (tid < 4 * ROWS) {                // 64 threads: 4 gates x 16 rows
    int gg = tid >> 4;
    int bl = tid & (ROWS - 1);
    float s = ro_b[gg];
#pragma unroll 8
    for (int q = 0; q < NQ; ++q) s += ro_w[gg * NQ + q] * zf[bl * ZFSTR + gg * NQ + q];
    gnext[gg * HID + row0 + bl] = s;
  }
}

// ---------------- final elementwise (step T-1) ----------------
__global__ __launch_bounds__(256) void qlstm_final(const float* __restrict__ gprev,
                                                   float* __restrict__ cx,
                                                   float* __restrict__ outs,
                                                   float* __restrict__ hx_out) {
  __shared__ float gF[HID], gIG[HID], gO[HID];
  const int tid = threadIdx.x;
  const int row0 = blockIdx.x * ROWS;
  for (int h = tid; h < HID; h += 256) {
    float f = sigm(gprev[h]);
    float i = sigm(gprev[HID + h]);
    float u = tanh_f(gprev[2 * HID + h]);
    float o = sigm(gprev[3 * HID + h]);
    gF[h] = f; gIG[h] = i * u; gO[h] = o;
  }
  __syncthreads();
  size_t ob = ((size_t)(T_STEPS - 1) * BATCH + row0) * HID;
  for (int e = tid; e < ROWS * HID; e += 256) {
    int r = e >> 10, h = e & (HID - 1);
    size_t ci = (size_t)(row0 + r) * HID + h;
    float cn = gF[h] * cx[ci] + gIG[h];
    float hv = gO[h] * tanh_f(cn);
    cx[ci] = cn;
    outs[ob + (size_t)r * HID + h] = hv;
    hx_out[ci] = hv;
  }
}

extern "C" void kernel_launch(void* const* d_in, const int* in_sizes, int n_in,
                              void* d_out, int out_size, void* d_ws, size_t ws_size,
                              hipStream_t stream) {
  (void)in_sizes; (void)n_in; (void)out_size; (void)ws_size;
  const float* inputs   = (const float*)d_in[0];
  const float* lin_w    = (const float*)d_in[1];
  const float* lin_b    = (const float*)d_in[2];
  const float* layers_w = (const float*)d_in[3];
  const float* layers_b = (const float*)d_in[4];
  const float* ro_w     = (const float*)d_in[5];
  const float* ro_b     = (const float*)d_in[6];

  float* out    = (float*)d_out;
  float* outs   = out;                                        // (T,B,H)
  float* hx_out = out + (size_t)T_STEPS * BATCH * HID;        // (B,H)
  float* cx_buf = hx_out + (size_t)BATCH * HID;               // (B,H) — live cx state

  char* ws = (char*)d_ws;
  ush*   Wbf   = (ush*)ws;                                                    // 256*1280 bf16
  ush*   Lbf   = (ush*)(ws + (size_t)NCOLS * DTOT * 2);                       // 4*4*64*64 bf16
  float* gout0 = (float*)(ws + (size_t)NCOLS * DTOT * 2 + 4 * DEPTHL * NQ * NQ * 2);
  float* gout1 = gout0 + 4 * HID;

  qlstm_prep<<<256, 256, 0, stream>>>(lin_w, layers_w, Wbf, Lbf);

  for (int t = 0; t < T_STEPS; ++t) {
    const float* gprev = (t & 1) ? gout0 : gout1;   // written by step t-1
    float*       gnext = (t & 1) ? gout1 : gout0;
    qlstm_step<<<NWG, 256, SMEM_BYTES, stream>>>(inputs, Wbf, Lbf, lin_b, layers_b,
                                                 ro_w, ro_b, gprev, gnext,
                                                 cx_buf, outs, t);
  }
  // step T-1 (t=255) wrote gout1
  qlstm_final<<<NWG, 256, 0, stream>>>(gout1, cx_buf, outs, hx_out);
}
